// OHCNN_48971217109565
// MI455X (gfx1250) — matrix operand order repaired and checked by reading, not compiled, not run on hardware
//
#include <hip/hip_runtime.h>
#include <hip/hip_bf16.h>

typedef __attribute__((ext_vector_type(16))) __bf16 v16bf;
typedef __attribute__((ext_vector_type(8)))  float  v8f;

// Problem constants
#define BATCH 32
#define LPOS  100
#define KDIM  30000
#define NCOLS 1000
#define NCLS  104
#define MROWS (BATCH * LPOS)   // 3200

// GEMM tiling
#define BM 128
#define BN 64
#define BK 32
#define NKT ((KDIM + BK - 1) / BK)   // 938

// LDS layout in u32 (each u32 = packed bf16 pair for K=2p,2p+1); row stride 17 (coprime w/ 64 banks)
#define ROWU 17
#define A_HI_OFF 0
#define A_LO_OFF (BM * ROWU)
#define B_HI_OFF (2 * BM * ROWU)
#define B_LO_OFF (2 * BM * ROWU + BN * ROWU)
#define BUF_U    (2 * BM * ROWU + 2 * BN * ROWU)   // 6528 u32 per buffer

struct FragU { unsigned int u[8]; };

// Truncation split of two consecutive f32 into packed bf16-hi pair and bf16-lo pair.
// hi16(x) = top 16 bits (truncate); lo = top 16 bits of (x - hi).
// Packing two high halves = one v_perm_b32. Cost: ~3 VALU per element.
__device__ __forceinline__ void split_pack2(float a, float b,
                                            unsigned int& hi, unsigned int& lo) {
  unsigned int ua = __builtin_bit_cast(unsigned int, a);
  unsigned int ub = __builtin_bit_cast(unsigned int, b);
  // D[15:0] = a[31:16], D[31:16] = b[31:16]
  hi = __builtin_amdgcn_perm(ub, ua, 0x07060302u);
  float ra = a - __builtin_bit_cast(float, ua & 0xffff0000u);
  float rb = b - __builtin_bit_cast(float, ub & 0xffff0000u);
  unsigned int ura = __builtin_bit_cast(unsigned int, ra);
  unsigned int urb = __builtin_bit_cast(unsigned int, rb);
  lo = __builtin_amdgcn_perm(urb, ura, 0x07060302u);
}

__device__ __forceinline__ v8f wmma_bf16(const FragU& a, const FragU& b, v8f c) {
  v16bf av = __builtin_bit_cast(v16bf, a);
  v16bf bv = __builtin_bit_cast(v16bf, b);
  return __builtin_amdgcn_wmma_f32_16x16x32_bf16(false, av, false, bv,
                                                 (short)0, c, false, false);
}

// Kernel 1: h[3200,1000] = relu(x @ conv_w^T + conv_b), bf16x3 split GEMM
__global__ __launch_bounds__(256)
void gemm_relu_kernel(const float* __restrict__ X, const float* __restrict__ W,
                      const float* __restrict__ bias, float* __restrict__ H)
{
  __shared__ unsigned int lds[2 * BUF_U];   // ~51 KB, double buffered

  const int tid    = threadIdx.x;
  const int mblock = blockIdx.x * BM;       // 25 blocks, exact
  const int nblock = blockIdx.y * BN;       // 16 blocks, padded past 1000

  float4 pa[4];   // staged A chunk: 128x32 f32 / 256 thr = 4 float4
  float4 pb[2];   // staged B chunk:  64x32 f32 / 256 thr = 2 float4

  auto load_global = [&](int kt) {
    const int kbase = kt * BK;
    #pragma unroll
    for (int i = 0; i < 4; ++i) {
      int quad = tid + 256 * i;          // [0,1024)
      int r    = quad >> 3;              // tile row 0..127
      int gk   = kbase + ((quad & 7) << 2);
      const float* p = X + (size_t)(mblock + r) * KDIM + gk;
      float4 t = {0.f, 0.f, 0.f, 0.f};
      if (gk + 4 <= KDIM)      t = *(const float4*)p;
      else {
        if (gk + 0 < KDIM) t.x = p[0];
        if (gk + 1 < KDIM) t.y = p[1];
        if (gk + 2 < KDIM) t.z = p[2];
        if (gk + 3 < KDIM) t.w = p[3];
      }
      pa[i] = t;
    }
    #pragma unroll
    for (int i = 0; i < 2; ++i) {
      int quad = tid + 256 * i;          // [0,512)
      int r    = quad >> 3;              // tile n-row 0..63
      int gk   = kbase + ((quad & 7) << 2);
      int gn   = nblock + r;
      float4 t = {0.f, 0.f, 0.f, 0.f};
      if (gn < NCOLS) {
        const float* p = W + (size_t)gn * KDIM + gk;
        if (gk + 4 <= KDIM)    t = *(const float4*)p;
        else {
          if (gk + 0 < KDIM) t.x = p[0];
          if (gk + 1 < KDIM) t.y = p[1];
          if (gk + 2 < KDIM) t.z = p[2];
          if (gk + 3 < KDIM) t.w = p[3];
        }
      }
      pb[i] = t;
    }
  };

  auto store_lds = [&](int buf) {
    unsigned int* base = lds + buf * BUF_U;
    #pragma unroll
    for (int i = 0; i < 4; ++i) {
      int quad = tid + 256 * i;
      int r    = quad >> 3;
      int p0   = (quad & 7) << 1;        // pair index within row
      unsigned int h01, l01, h23, l23;
      split_pack2(pa[i].x, pa[i].y, h01, l01);
      split_pack2(pa[i].z, pa[i].w, h23, l23);
      base[A_HI_OFF + r * ROWU + p0]     = h01;
      base[A_HI_OFF + r * ROWU + p0 + 1] = h23;
      base[A_LO_OFF + r * ROWU + p0]     = l01;
      base[A_LO_OFF + r * ROWU + p0 + 1] = l23;
    }
    #pragma unroll
    for (int i = 0; i < 2; ++i) {
      int quad = tid + 256 * i;
      int r    = quad >> 3;
      int p0   = (quad & 7) << 1;
      unsigned int h01, l01, h23, l23;
      split_pack2(pb[i].x, pb[i].y, h01, l01);
      split_pack2(pb[i].z, pb[i].w, h23, l23);
      base[B_HI_OFF + r * ROWU + p0]     = h01;
      base[B_HI_OFF + r * ROWU + p0 + 1] = h23;
      base[B_LO_OFF + r * ROWU + p0]     = l01;
      base[B_LO_OFF + r * ROWU + p0 + 1] = l23;
    }
  };

  const int lane   = tid & 31;
  const int waveId = tid >> 5;
  const int wm     = (waveId & 3) << 5;   // wave M offset (0,32,64,96)
  const int wn     = (waveId >> 2) << 5;  // wave N offset (0,32)
  const int ml     = lane & 15;
  const int half   = lane >> 4;

  v8f acc[2][2];
  #pragma unroll
  for (int mt = 0; mt < 2; ++mt)
    #pragma unroll
    for (int nt = 0; nt < 2; ++nt)
      acc[mt][nt] = (v8f){0.f,0.f,0.f,0.f,0.f,0.f,0.f,0.f};

  auto compute = [&](int buf) {
    const unsigned int* base = lds + buf * BUF_U;
    // A fragment pair indices per ISA 16-bit A 16x32 layout:
    // lanes 0-15:  v0..3 -> K pairs 0..3,  v4..7 -> pairs 8..11
    // lanes 16-31: v0..3 -> K pairs 4..7,  v4..7 -> pairs 12..15
    FragU ahi[2], alo[2];
    #pragma unroll
    for (int mt = 0; mt < 2; ++mt) {
      int row = wm + (mt << 4) + ml;
      const unsigned int* rh = base + A_HI_OFF + row * ROWU;
      const unsigned int* rl = base + A_LO_OFF + row * ROWU;
      #pragma unroll
      for (int v = 0; v < 8; ++v) {
        int p = ((v < 4) ? v : v + 4) + (half << 2);
        ahi[mt].u[v] = rh[p];
        alo[mt].u[v] = rl[p];
      }
    }
    // B fragment per ISA B 32x16 layout: lane n = lane%16;
    // same K pair order as A: lanes 0-15 pairs 0..3,8..11; lanes 16-31 pairs 4..7,12..15
    FragU bhi[2], blo[2];
    #pragma unroll
    for (int nt = 0; nt < 2; ++nt) {
      int row = wn + (nt << 4) + ml;
      const unsigned int* rh = base + B_HI_OFF + row * ROWU;
      const unsigned int* rl = base + B_LO_OFF + row * ROWU;
      #pragma unroll
      for (int v = 0; v < 8; ++v) {
        int p = ((v < 4) ? v : v + 4) + (half << 2);
        bhi[nt].u[v] = rh[p];
        blo[nt].u[v] = rl[p];
      }
    }
    // bf16x3: hi*hi + hi*lo + lo*hi, f32 accumulate
    #pragma unroll
    for (int mt = 0; mt < 2; ++mt)
      #pragma unroll
      for (int nt = 0; nt < 2; ++nt) {
        v8f c = acc[mt][nt];
        c = wmma_bf16(ahi[mt], bhi[nt], c);
        c = wmma_bf16(ahi[mt], blo[nt], c);
        c = wmma_bf16(alo[mt], bhi[nt], c);
        acc[mt][nt] = c;
      }
  };

  load_global(0);
  store_lds(0);
  __syncthreads();
  for (int kt = 0; kt < NKT; ++kt) {
    int cur = kt & 1;
    if (kt + 1 < NKT) load_global(kt + 1);
    compute(cur);
    if (kt + 1 < NKT) {
      store_lds(cur ^ 1);
      __syncthreads();
    }
  }

  // Epilogue: bias + ReLU, store to H[row*NCOLS + co]
  #pragma unroll
  for (int nt = 0; nt < 2; ++nt) {
    int gco = nblock + wn + (nt << 4) + ml;
    if (gco >= NCOLS) continue;
    float bv = bias[gco];
    #pragma unroll
    for (int mt = 0; mt < 2; ++mt) {
      v8f c = acc[mt][nt];
      #pragma unroll
      for (int r = 0; r < 8; ++r) {
        int grow = mblock + wm + (mt << 4) + r + (half << 3);
        float v = c[r] + bv;
        H[(size_t)grow * NCOLS + gco] = v > 0.f ? v : 0.f;
      }
    }
  }
}

// Kernel 2: pooled[b, co*10+g] = mean_{j<10} h[b*100 + g*10+j, co]
__global__ __launch_bounds__(256)
void pool_kernel(const float* __restrict__ H, float* __restrict__ P)
{
  int id = blockIdx.x * blockDim.x + threadIdx.x;
  if (id >= BATCH * NCOLS) return;
  int b  = id / NCOLS;
  int co = id % NCOLS;
  const float* base = H + (size_t)b * LPOS * NCOLS + co;
  #pragma unroll
  for (int g = 0; g < 10; ++g) {
    float s = 0.f;
    #pragma unroll
    for (int j = 0; j < 10; ++j) s += base[(size_t)(g * 10 + j) * NCOLS];
    P[(size_t)b * 10000 + co * 10 + g] = s * 0.1f;
  }
}

// Kernel 3: per-batch normalize + FC (deterministic reductions, no atomics)
__global__ __launch_bounds__(256)
void norm_fc_kernel(const float* __restrict__ P, const float* __restrict__ FW,
                    const float* __restrict__ FB, float* __restrict__ OUT)
{
  __shared__ float v[10000];
  __shared__ float red[256];
  const int b = blockIdx.x;
  const int t = threadIdx.x;
  const float* p = P + (size_t)b * 10000;

  float ss = 0.f;
  for (int j = t; j < 10000; j += 256) { float x = p[j]; v[j] = x; ss += x * x; }
  red[t] = ss;
  __syncthreads();
  for (int s = 128; s > 0; s >>= 1) {
    if (t < s) red[t] += red[t + s];
    __syncthreads();
  }
  const float inv = 1.f / sqrtf(1.f + red[0]);

  const int wave = t >> 5;
  const int lane = t & 31;
  for (int c = wave; c < NCLS; c += 8) {
    const float* w = FW + (size_t)c * 10000;
    float d = 0.f;
    for (int j = lane; j < 10000; j += 32) d += v[j] * w[j];
    #pragma unroll
    for (int off = 16; off > 0; off >>= 1) d += __shfl_down(d, off, 32);
    if (lane == 0) OUT[b * NCLS + c] = d * inv + FB[c];
  }
}

extern "C" void kernel_launch(void* const* d_in, const int* in_sizes, int n_in,
                              void* d_out, int out_size, void* d_ws, size_t ws_size,
                              hipStream_t stream) {
  const float* x      = (const float*)d_in[0];   // [32,100,30000]
  const float* conv_w = (const float*)d_in[1];   // [1000,30000]
  const float* conv_b = (const float*)d_in[2];   // [1000]
  const float* fc_w   = (const float*)d_in[3];   // [104,10000]
  const float* fc_b   = (const float*)d_in[4];   // [104]
  float* out = (float*)d_out;                    // [32,104]

  float* H = (float*)d_ws;                       // [3200,1000] = 12.8 MB
  float* P = H + (size_t)MROWS * NCOLS;          // [32,10000]  = 1.28 MB

  dim3 g1(MROWS / BM, (NCOLS + BN - 1) / BN);    // 25 x 16
  gemm_relu_kernel<<<g1, 256, 0, stream>>>(x, conv_w, conv_b, H);

  int nPool = BATCH * NCOLS;
  pool_kernel<<<(nPool + 255) / 256, 256, 0, stream>>>(H, P);

  norm_fc_kernel<<<BATCH, 256, 0, stream>>>(P, fc_w, fc_b, out);
}
